// DSTDGC_20057497272940
// MI455X (gfx1250) — matrix operand order repaired and checked by reading, not compiled, not run on hardware
//
#include <hip/hip_runtime.h>

// ---------------------------------------------------------------------------
// DSTDGC fused pipeline for MI455X (gfx1250, wave32, WMMA bf16 16x16x32)
// N=32, CIN=COUT=64, RED=2, T=64, V=128
//
// Roofline: ~15 GFLOP matrix work + 67M tanh, ~200-270MB true HBM traffic
// (192MB L2 keeps x / xm resident across kernels) -> ~10us bandwidth bound;
// WMMA bf16 (f32 accumulate) keeps the GEMMs off the f32-VALU path.
//
// All WMMA operands live in LDS "fragment pools" [frag][lane][16 bf16] so a
// fragment load is 2x ds_load_b128 per lane (no scalar u16 gathers).
// Bulk global->LDS staging goes through the Tensor Data Mover (6-arg
// clang-23 builtin; D# built per CDNA5 ISA).
// ---------------------------------------------------------------------------

typedef __attribute__((ext_vector_type(16))) __bf16 v16bf;
typedef __attribute__((ext_vector_type(8)))  float  v8f;

__device__ __forceinline__ float fast_tanh(float v) {
#if __has_builtin(__builtin_amdgcn_tanhf)
  return __builtin_amdgcn_tanhf(v);   // v_tanh_f32 (CDNA5 transcendental)
#else
  return tanhf(v);
#endif
}

// ISA 7.12.2 16-bit A-matrix element map: lane half selects K octets.
__device__ __forceinline__ int a_kmap(int half, int e) {
  return (e < 8) ? (half * 8 + e) : (16 + half * 8 + (e - 8));
}

// Fragment pool: pool[fragIdx][lane][16 bf16], 32B per lane -> 2x b128.
__device__ __forceinline__ v16bf load_frag(const __bf16* pool, int fragIdx,
                                           int lane) {
  const uint4* p = (const uint4*)(pool + ((size_t)fragIdx * 32 + lane) * 16);
  union { v16bf v; uint4 q[2]; } f;
  f.q[0] = p[0];
  f.q[1] = p[1];
  return f.v;
}

__device__ __forceinline__ void store_frag_half(__bf16* dst, v16bf v) {
  union { v16bf v; uint4 q[2]; } f;
  f.v = v;
  uint4* p = (uint4*)dst;
  p[0] = f.q[0];
  p[1] = f.q[1];
}

__device__ __forceinline__ v8f wmma_bf16(v16bf a, v16bf b, v8f c) {
  return __builtin_amdgcn_wmma_f32_16x16x32_bf16(
      false, a, false, b, (short)0, c, false, false);
}

// ---------------------------------------------------------------------------
// Tensor Data Mover: build D# (groups 0/1) per CDNA5 ISA
// and issue a 2D tile load: tile_d0 x tile_d1 elements of (1<<dsz_log) bytes,
// global rows strided by row_stride elements; packed contiguously into LDS.
// Toolchain probe: this environment's builtin is the 6-arg clang-23 form
// (uint32x4, int32x8, int32x4, int32x4, int32x8, i32 cpol).
// ---------------------------------------------------------------------------
#if __has_builtin(__builtin_amdgcn_tensor_load_to_lds) && \
    __has_builtin(__builtin_amdgcn_s_wait_tensorcnt)
#define HAVE_TDM 1
typedef __attribute__((ext_vector_type(4))) unsigned int v4u;
typedef __attribute__((ext_vector_type(8))) int v8i;
typedef __attribute__((ext_vector_type(4))) int v4i;

__device__ __forceinline__ void tdm_load_2d(void* lds_dst, const void* gsrc,
                                            unsigned dsz_log,
                                            unsigned tile_d0, unsigned tile_d1,
                                            unsigned tensor_d0,
                                            unsigned tensor_d1,
                                            unsigned row_stride) {
  const unsigned lds_addr = (unsigned)(size_t)lds_dst;  // low 32b = LDS offset
  const unsigned long long ga = (unsigned long long)(size_t)gsrc;
  v4u g0;
  g0.x = 1u;                                          // count=1, user mode
  g0.y = lds_addr;                                    // [63:32] lds_addr
  g0.z = (unsigned)(ga & 0xffffffffu);                // [95:64] global lo
  g0.w = (unsigned)((ga >> 32) & 0x01ffffffu)         // [120:96] global hi
         | (2u << 30);                                // [127:126] type=2
  v8i g1;
  g1[0] = (int)(dsz_log << 16);            // wg_mask=0, data_size, no flags
  g1[1] = (int)(tensor_d0 << 16);          // [79:48] tensor_dim0 (lo16)
  g1[2] = (int)((tensor_d0 >> 16) | (tensor_d1 << 16));  // dim0 hi / dim1 lo
  g1[3] = (int)((tensor_d1 >> 16) | (tile_d0 << 16));    // dim1 hi / tile_d0
  g1[4] = (int)(tile_d1 & 0xffffu);        // tile_dim1, tile_dim2=0
  g1[5] = (int)row_stride;                 // [207:160] tensor_dim0_stride lo
  g1[6] = 0;                               // stride hi, dim1_stride lo
  g1[7] = 0;                               // dim1_stride hi
  const v4i gz4 = {0, 0, 0, 0};            // groups 2/3 unused (<=2D tensor)
  const v8i gz8 = {0, 0, 0, 0, 0, 0, 0, 0};
  __builtin_amdgcn_tensor_load_to_lds(g0, g1, gz4, gz4, gz8, 0);
}
#endif

// ---------------------------------------------------------------------------
// K1: a[n, r*64+t, v] = Wm1[r,:].x[n,:,t,v] + bm1[r]   (likewise b with Wm2)
// Streaming GEMV (K=64, M=4): memory bound, VALU is fine.
// ---------------------------------------------------------------------------
__global__ void __launch_bounds__(128)
k1_proj_ab(const float* __restrict__ x,
           const float* __restrict__ Wm1, const float* __restrict__ bm1,
           const float* __restrict__ Wm2, const float* __restrict__ bm2,
           float* __restrict__ a_ws, float* __restrict__ b_ws) {
  const int nt = blockIdx.x;
  const int n = nt >> 6, t = nt & 63;
  const int v = threadIdx.x;                  // 0..127

  __shared__ float wsh[4][64];
  {
    const int i = threadIdx.x;
    wsh[(i >> 6) + 0][i & 63] = Wm1[i];
    wsh[(i >> 6) + 2][i & 63] = Wm2[i];
  }
  __syncthreads();

  const float* xp = x + (((size_t)n * 64) * 64 + t) * 128 + v;  // c = 0
  float a0 = 0.f, a1 = 0.f, b0 = 0.f, b1 = 0.f;
#pragma unroll 8
  for (int c = 0; c < 64; ++c) {
    const float xv = xp[(size_t)c * 64 * 128];
    a0 += wsh[0][c] * xv;  a1 += wsh[1][c] * xv;
    b0 += wsh[2][c] * xv;  b1 += wsh[3][c] * xv;
  }
  const size_t base = (size_t)n * 128 * 128;
  a_ws[base + (size_t)(t)      * 128 + v] = a0 + bm1[0];
  a_ws[base + (size_t)(64 + t) * 128 + v] = a1 + bm1[1];
  b_ws[base + (size_t)(t)      * 128 + v] = b0 + bm2[0];
  b_ws[base + (size_t)(64 + t) * 128 + v] = b1 + bm2[1];
}

// ---------------------------------------------------------------------------
// K2: xm[n,o,v,w] = sum_k Wrm[o,k]*tanh(a[n,k,v]-b[n,k,w]) + brm[o] + A[o,v,w]
// One block per (n, 16x16 (v,w)-tile); WMMA M=64, N=256, K=128.
// a/b columns staged by TDM 2D tile loads (16 x 128 f32, row stride 512B).
// Output written to global ALREADY in B-fragment layout per (n,o) so K3
// stages it with a single contiguous copy.
// grid = (64 tilepairs, 32 n), 256 threads = 8 waves.
// ---------------------------------------------------------------------------
__global__ void __launch_bounds__(256)
k2_xm(const float* __restrict__ a_ws, const float* __restrict__ b_ws,
      const float* __restrict__ Wrm, const float* __restrict__ brm,
      const float* __restrict__ Ag, __bf16* __restrict__ xm_ws) {
  __shared__ float a_s[128][16];                       //  8 KB
  __shared__ float b_s[128][16];                       //  8 KB
  __shared__ __align__(16) __bf16 wpool[4 * 4 * 32 * 16];   // 16 KB Wrm A-frags
  __shared__ __align__(16) __bf16 dpool[4 * 16 * 32 * 16];  // 64 KB d B-frags
  __shared__ float brm_s[64];

  const int n   = blockIdx.y;
  const int vt  = blockIdx.x >> 3;                     // v-tile 0..7
  const int wt  = blockIdx.x & 7;                      // w-tile 0..7
  const int tid = threadIdx.x;

#ifdef HAVE_TDM
  // TDM: 2D tile 16 cols x 128 rows of f32 out of a 128-stride tensor.
  if (tid < 32) {
    tdm_load_2d(&a_s[0][0], a_ws + (size_t)n * 16384 + vt * 16,
                /*dsz=4B*/2, /*tile*/16, 128, /*tensor*/128, 128, /*stride*/128);
    tdm_load_2d(&b_s[0][0], b_ws + (size_t)n * 16384 + wt * 16,
                2, 16, 128, 128, 128, 128);
  }
#else
  for (int i = tid; i < 128 * 16; i += 256) {
    const int k = i >> 4, j = i & 15;
    a_s[k][j] = a_ws[((size_t)n * 128 + k) * 128 + vt * 16 + j];
    b_s[k][j] = b_ws[((size_t)n * 128 + k) * 128 + wt * 16 + j];
  }
#endif
  // Wrm A-fragment pool: frag (mt,ks): elem e of lane (half,m) is
  // Wrm[mt*16+m][ks*32 + a_kmap(half,e)].  (Overlaps with the TDM DMA.)
  for (int idx = tid; idx < 4 * 4 * 32 * 16; idx += 256) {
    const int e = idx & 15, lane_ = (idx >> 4) & 31;
    const int ks = (idx >> 9) & 3, mt = idx >> 11;
    const int m = lane_ & 15, half = lane_ >> 4;
    wpool[idx] = (__bf16)Wrm[(mt * 16 + m) * 128 + ks * 32 + a_kmap(half, e)];
  }
  if (tid < 64) brm_s[tid] = brm[tid];
#ifdef HAVE_TDM
  if (tid < 32) __builtin_amdgcn_s_wait_tensorcnt(0);
#endif
  __syncthreads();

  // d B-fragment pool: thread owns col = vloc*16 + wloc; packs 16 k-values
  // into a v16bf and stores 32B per (ks,half) block (2x ds_store_b128).
  {
    const int col = tid, nt = col >> 4, nl = col & 15;   // nt = vloc, nl = wloc
#pragma unroll
    for (int ks = 0; ks < 4; ++ks)
#pragma unroll
      for (int half = 0; half < 2; ++half) {
        v16bf pk;
#pragma unroll
        for (int i = 0; i < 16; ++i) {
          const int k = ks * 32 + a_kmap(half, i);
          pk[i] = (__bf16)fast_tanh(a_s[k][nt] - b_s[k][nl]);
        }
        store_frag_half(&dpool[((ks * 16 + nt) * 32 + half * 16 + nl) * 16], pk);
      }
  }
  __syncthreads();

  const int lane = tid & 31, wave = tid >> 5;
  const int ot = wave & 3;                        // o-tile (M)
  const int ng = (wave >> 2) * 8;                 // N-tile base (0 or 8)

  const v8f zero = {0.f, 0.f, 0.f, 0.f, 0.f, 0.f, 0.f, 0.f};
  v8f acc[8];
#pragma unroll
  for (int j = 0; j < 8; ++j) acc[j] = zero;

#pragma unroll
  for (int ks = 0; ks < 4; ++ks) {                // K = 128 = 4 x 32
    const v16bf af = load_frag(wpool, ot * 4 + ks, lane);
#pragma unroll
    for (int j = 0; j < 8; ++j) {
      const v16bf bfr = load_frag(dpool, ks * 16 + (ng + j), lane);
      acc[j] = wmma_bf16(af, bfr, acc[j]);
    }
  }

  // Epilogue: + brm[o] + A[o,v,w]; pack 8 bf16 per row -> 16B global stores
  // directly into the B-fragment layout K3 consumes.
  const int wloc   = lane & 15;
  const int rowadd = (lane >> 4) * 8;
  const int kv = vt >> 1, halfv = vt & 1;
#pragma unroll
  for (int r = 0; r < 8; ++r) {
    const int o = ot * 16 + rowadd + r;
    union { uint4 q; __bf16 h[8]; } p;
#pragma unroll
    for (int j = 0; j < 8; ++j) {
      const int v = vt * 16 + ng + j;
      const int w = wt * 16 + wloc;
      p.h[j] = (__bf16)(acc[j][r] + brm_s[o] +
                        Ag[((size_t)o * 128 + v) * 128 + w]);
    }
    const size_t elem =
        (((((size_t)n * 64 + o) * 4 + kv) * 8 + wt) * 32 + halfv * 16 + wloc) *
            16 + ng;
    *(uint4*)(xm_ws + elem) = p.q;               // 16B aligned (ng in {0,8})
  }
}

// ---------------------------------------------------------------------------
// K3: recompute xf[n,:,t,:] in-LDS (WMMA, K=64), then
//     out[n,c,t,w] = sum_v xf[c,v] * xm[n,t,v,w]   (WMMA, K=128)
// xm arrives pre-swizzled in fragment layout -> one 32KB TDM 1D tile load.
// grid = N*T blocks, 128 threads = 4 waves (wave == M-tile).
// ---------------------------------------------------------------------------
__global__ void __launch_bounds__(128)
k3_out(const float* __restrict__ x,
       const float* __restrict__ Wf, const float* __restrict__ bfv,
       const __bf16* __restrict__ xm_ws, float* __restrict__ out) {
  __shared__ __align__(16) __bf16 xspool[2 * 8 * 32 * 16];  // 16 KB x B-frags
  __shared__ __align__(16) __bf16 wfpool[4 * 2 * 32 * 16];  //  8 KB Wf A-frags
  __shared__ __align__(16) __bf16 xfpool[4 * 4 * 32 * 16];  // 16 KB xf A-frags
  __shared__ __align__(16) __bf16 xmpool[4 * 8 * 32 * 16];  // 32 KB xm B-frags
  __shared__ float bf_s[64];

  const int ntb = blockIdx.x;
  const int n = ntb >> 6, t = ntb & 63;
  const int tid = threadIdx.x;                    // 0..127
  const int lane = tid & 31, wave = tid >> 5;

#ifdef HAVE_TDM
  // TDM: contiguous 32KB slab = 1D tile of 4096 x 8B.
  if (tid < 32)
    tdm_load_2d(xmpool, xm_ws + (size_t)(n * 64 + t) * 16384,
                /*dsz=8B*/3, /*tile*/4096, 1, /*tensor*/4096, 1, /*stride*/4096);
#endif

  // x slice -> B-fragment pool. Thread owns v; coalesced global reads, packs
  // 16 c (a_kmap order) per (ks,half) block, 2x ds_store_b128 each.
  {
    const int v = tid, nt = v >> 4, nl = v & 15;
    const float* xp = x + (((size_t)n * 64) * 64 + t) * 128 + v;  // c = 0
#pragma unroll
    for (int ks = 0; ks < 2; ++ks)
#pragma unroll
      for (int half = 0; half < 2; ++half) {
        v16bf pk;
#pragma unroll
        for (int i = 0; i < 16; ++i) {
          const int c = ks * 32 + a_kmap(half, i);
          pk[i] = (__bf16)xp[(size_t)c * 64 * 128];
        }
        store_frag_half(&xspool[((ks * 8 + nt) * 32 + half * 16 + nl) * 16], pk);
      }
  }
  // Wf A-fragment pool.
  for (int idx = tid; idx < 4 * 2 * 32 * 16; idx += 128) {
    const int e = idx & 15, lane_ = (idx >> 4) & 31;
    const int ks = (idx >> 9) & 1, mt = idx >> 10;
    const int m = lane_ & 15, half = lane_ >> 4;
    wfpool[idx] = (__bf16)Wf[(mt * 16 + m) * 64 + ks * 32 + a_kmap(half, e)];
  }
  if (tid < 64) bf_s[tid] = bfv[tid];
#ifdef HAVE_TDM
  if (tid < 32) __builtin_amdgcn_s_wait_tensorcnt(0);
#else
  {
    const uint4* src = (const uint4*)(xm_ws + (size_t)(n * 64 + t) * 16384);
    uint4* dst = (uint4*)xmpool;
    for (int i = tid; i < 2048; i += 128) dst[i] = src[i];
  }
#endif
  __syncthreads();

  const v8f zero = {0.f, 0.f, 0.f, 0.f, 0.f, 0.f, 0.f, 0.f};
  const int cloc   = lane & 15;
  const int rowadd = (lane >> 4) * 8;

  // Phase A: xf = Wf (64x64) * xs (64x128); scatter C-frags into A-frag pool.
  {
    v8f acc[8];
#pragma unroll
    for (int j = 0; j < 8; ++j) acc[j] = zero;
#pragma unroll
    for (int ks = 0; ks < 2; ++ks) {
      const v16bf af = load_frag(wfpool, wave * 2 + ks, lane);
#pragma unroll
      for (int j = 0; j < 8; ++j) {
        const v16bf bfr = load_frag(xspool, ks * 8 + j, lane);
        acc[j] = wmma_bf16(af, bfr, acc[j]);
      }
    }
    // (o = wave*16+rowadd+r, v = j*16+cloc) -> A-frag (mt=wave, ks=j>>1),
    // lane' = (j&1)*16 + (o&15), e = cloc.
#pragma unroll
    for (int j = 0; j < 8; ++j)
#pragma unroll
      for (int r = 0; r < 8; ++r) {
        const int o = wave * 16 + rowadd + r;
        const int idx =
            ((wave * 4 + (j >> 1)) * 32 + (j & 1) * 16 + rowadd + r) * 16 +
            cloc;
        xfpool[idx] = (__bf16)(acc[j][r] + bf_s[o]);
      }
  }
  __syncthreads();

  // Phase B: out = xf (64x128) * xm (128x128)
  {
    v8f acc[8];
#pragma unroll
    for (int j = 0; j < 8; ++j) acc[j] = zero;
#pragma unroll
    for (int ks = 0; ks < 4; ++ks) {              // K = 128
      const v16bf af = load_frag(xfpool, wave * 4 + ks, lane);
#pragma unroll
      for (int j = 0; j < 8; ++j) {
        const v16bf bfr = load_frag(xmpool, ks * 8 + j, lane);
        acc[j] = wmma_bf16(af, bfr, acc[j]);
      }
    }
#pragma unroll
    for (int j = 0; j < 8; ++j)
#pragma unroll
      for (int r = 0; r < 8; ++r) {
        const int c = wave * 16 + rowadd + r;
        out[(((size_t)n * 64 + c) * 64 + t) * 128 + j * 16 + cloc] = acc[j][r];
      }
  }
}

// ---------------------------------------------------------------------------
// Launcher. Workspace layout (needs ~71.3 MB):
//   [0,   2MB) a_ws   f32 [32][128][128]
//   [2MB, 4MB) b_ws   f32 [32][128][128]
//   [4MB, 68MB) xm_ws bf16 [32][64] x 16384 (fragment layout per (n,o))
// ---------------------------------------------------------------------------
extern "C" void kernel_launch(void* const* d_in, const int* in_sizes, int n_in,
                              void* d_out, int out_size, void* d_ws, size_t ws_size,
                              hipStream_t stream) {
  const float* x   = (const float*)d_in[0];
  const float* A   = (const float*)d_in[1];
  const float* Wf  = (const float*)d_in[2];
  const float* bf  = (const float*)d_in[3];
  const float* Wm1 = (const float*)d_in[4];
  const float* bm1 = (const float*)d_in[5];
  const float* Wm2 = (const float*)d_in[6];
  const float* bm2 = (const float*)d_in[7];
  const float* Wrm = (const float*)d_in[8];
  const float* brm = (const float*)d_in[9];
  float* out = (float*)d_out;

  char* ws = (char*)d_ws;
  float*  a_ws  = (float*)(ws);
  float*  b_ws  = (float*)(ws + (size_t)(2 << 20));
  __bf16* xm_ws = (__bf16*)(ws + (size_t)(4 << 20));

  k1_proj_ab<<<32 * 64, 128, 0, stream>>>(x, Wm1, bm1, Wm2, bm2, a_ws, b_ws);
  k2_xm<<<dim3(64, 32), 256, 0, stream>>>(a_ws, b_ws, Wrm, brm, A, xm_ws);
  k3_out<<<32 * 64, 128, 0, stream>>>(x, Wf, bf, xm_ws, out);
}
